// HorizontalGAT_39608188403995
// MI455X (gfx1250) — matrix operand order repaired and checked by reading, not compiled, not run on hardware
//
#include <hip/hip_runtime.h>
#include <math.h>

#define B_ 16
#define W_ 512
#define D_ 512
#define H_ 256
#define WP_ (W_ + 2)   // padded rows for conv (zero row before/after)
#define ALPHA_ 0.1f

typedef __attribute__((ext_vector_type(16))) _Float16 v16h;
typedef __attribute__((ext_vector_type(8)))  float    v8f;
typedef _Float16 half_t;

// ---------------------------------------------------------------------------
// WMMA helpers (CDNA5 wave32, v_wmma_f32_16x16x32_f16, D = A(16x32) B(32x16) + C)
// ---------------------------------------------------------------------------
__device__ __forceinline__ v8f wmma16(v16h a, v16h b, v8f c) {
  return __builtin_amdgcn_wmma_f32_16x16x32_f16(
      /*neg_a=*/false, a, /*neg_b=*/false, b,
      /*c_mod=*/(short)0, c, /*reuse_a=*/false, /*reuse_b=*/false);
}

// A (16x32, MxK) row-major, lda in halves.
// lane m=lane&15, g=lane>>4; elem e -> K = g*16+e
__device__ __forceinline__ v16h load_a_rowmajor(const half_t* A, int lda) {
  int lane = threadIdx.x & 31;
  int m = lane & 15, g = lane >> 4;
  const half_t* p = A + (size_t)m * lda + g * 16;
  v16h f;
#pragma unroll
  for (int i = 0; i < 8; ++i) { f[i] = p[i]; f[i + 8] = p[8 + i]; }
  return f;
}

// B (32x16, KxN) supplied as BT row-major [N x K] (contiguous along K).
// lane n=lane&15, g=lane>>4; elem e -> K = g*16+e
__device__ __forceinline__ v16h load_b_nmajor(const half_t* BT, int ldb) {
  int lane = threadIdx.x & 31;
  int n = lane & 15, g = lane >> 4;
  const half_t* p = BT + (size_t)n * ldb + g * 16;
  v16h f;
#pragma unroll
  for (int i = 0; i < 16; ++i) f[i] = p[i];
  return f;
}

// ---------------------------------------------------------------------------
// Conversion / layout kernels
// ---------------------------------------------------------------------------
__global__ void k_cvt(const float* __restrict__ src, half_t* __restrict__ dst, int n) {
  int i = blockIdx.x * blockDim.x + threadIdx.x;
  if (i < n) dst[i] = (half_t)src[i];
}

// x [B][W][D] f32 -> xh [B][W+2][D] f16 with zero pad rows (row 0 and W+1)
__global__ void k_cvt_xpad(const float* __restrict__ x, half_t* __restrict__ xh) {
  int idx = blockIdx.x * blockDim.x + threadIdx.x;
  int total = B_ * WP_ * D_;
  if (idx >= total) return;
  int d = idx % D_;
  int t = idx / D_;
  int row = t % WP_;
  int b = t / WP_;
  half_t v = (half_t)0.0f;
  if (row >= 1 && row <= W_)
    v = (half_t)x[((size_t)b * W_ + row - 1) * D_ + d];
  xh[idx] = v;
}

// conv weights [O,I,3] f32 -> wt[c][dk][o][i] f16 (BT layout per slice)
__global__ void k_cvt_convw(const float* __restrict__ wq, const float* __restrict__ wk,
                            const float* __restrict__ wv, const float* __restrict__ wp,
                            half_t* __restrict__ wt) {
  int idx = blockIdx.x * blockDim.x + threadIdx.x;
  int total = 4 * 3 * D_ * D_;
  if (idx >= total) return;
  int i = idx % D_;
  int t = idx / D_;
  int o = t % D_; t /= D_;
  int dk = t % 3;
  int c = t / 3;
  const float* w = (c == 0) ? wq : (c == 1) ? wk : (c == 2) ? wv : wp;
  wt[idx] = (half_t)w[(size_t)(o * D_ + i) * 3 + dk];
}

// src [B][R][C] f16 -> dst [B][C][R] f16 (coalesced writes: r fastest)
__global__ void k_transpose(const half_t* __restrict__ src, half_t* __restrict__ dst,
                            int R, int C) {
  int idx = blockIdx.x * blockDim.x + threadIdx.x;
  int total = B_ * R * C;
  if (idx >= total) return;
  int r = idx % R;
  int t = idx / R;
  int c = t % C;
  int b = t / C;
  dst[((size_t)b * C + c) * R + r] = src[((size_t)b * R + r) * C + c];
}

// ---------------------------------------------------------------------------
// Conv1d(k=3,pad=1) on zero-padded input: one wave -> 32x64 output block.
// All A loads unconditional (no exec-mask juggling in the K loop).
// grid: (W/32, D/64, B*4)  block: 32
// ---------------------------------------------------------------------------
__global__ void k_conv(const half_t* __restrict__ xh, const half_t* __restrict__ wt,
                       const float* __restrict__ bq, const float* __restrict__ bk,
                       const float* __restrict__ bv, const float* __restrict__ bp,
                       half_t* __restrict__ qh, half_t* __restrict__ kh,
                       half_t* __restrict__ vh, half_t* __restrict__ ph) {
  int wm = blockIdx.x, on = blockIdx.y, z = blockIdx.z;
  int c = z & 3, b = z >> 2;
  const half_t* xb = xh + (size_t)b * WP_ * D_;
  const half_t* wc = wt + (size_t)c * 3 * D_ * D_;
  const float* bias = (c == 0) ? bq : (c == 1) ? bk : (c == 2) ? bv : bp;
  half_t* out = ((c == 0) ? qh : (c == 1) ? kh : (c == 2) ? vh : ph) + (size_t)b * W_ * D_;

  v8f acc[2][4] = {};
#pragma unroll
  for (int dk = 0; dk < 3; ++dk) {
    int row0p = wm * 32 + dk;  // padded row of m=0 (global row wm*32+dk-1, +1 pad)
    const half_t* A0 = xb + (size_t)row0p * D_;
    const half_t* A1 = A0 + (size_t)16 * D_;
    const half_t* wdk = wc + (size_t)dk * D_ * D_ + (size_t)(on * 64) * D_;
    for (int kt = 0; kt < D_ / 32; ++kt) {
      __builtin_prefetch(wdk + (kt + 1) * 32, 0, 3);
      v16h a0 = load_a_rowmajor(A0 + kt * 32, D_);
      v16h a1 = load_a_rowmajor(A1 + kt * 32, D_);
      v16h bf[4];
#pragma unroll
      for (int ni = 0; ni < 4; ++ni)
        bf[ni] = load_b_nmajor(wdk + (size_t)(ni * 16) * D_ + kt * 32, D_);
#pragma unroll
      for (int ni = 0; ni < 4; ++ni) {
        acc[0][ni] = wmma16(a0, bf[ni], acc[0][ni]);
        acc[1][ni] = wmma16(a1, bf[ni], acc[1][ni]);
      }
    }
  }
  int lane = threadIdx.x & 31;
  int n = lane & 15, g = lane >> 4;
#pragma unroll
  for (int mi = 0; mi < 2; ++mi)
#pragma unroll
    for (int ni = 0; ni < 4; ++ni) {
      int col = on * 64 + ni * 16 + n;
      float bo = bias[col];
#pragma unroll
      for (int r = 0; r < 8; ++r) {
        int row = wm * 32 + mi * 16 + g * 8 + r;
        out[(size_t)row * D_ + col] = (half_t)(acc[mi][ni][r] + bo);
      }
    }
}

// ---------------------------------------------------------------------------
// 32x64-blocked GEMM: C[W x W] = A[W x D] . BT[W x D]^T (both N-major f16)
// grid: (W/32, W/64, B) block: 32.  MODE 0: raw f32 out. MODE 1: prior chain.
// ---------------------------------------------------------------------------
template <int MODE>
__global__ void k_qkgemm(const half_t* __restrict__ Ag, const half_t* __restrict__ BTg,
                         float* __restrict__ outg) {
  int wm = blockIdx.x, un = blockIdx.y, b = blockIdx.z;
  const half_t* A  = Ag  + (size_t)b * W_ * D_ + (size_t)(wm * 32) * D_;
  const half_t* BT = BTg + (size_t)b * W_ * D_ + (size_t)(un * 64) * D_;
  v8f acc[2][4] = {};
  for (int kt = 0; kt < D_ / 32; ++kt) {
    __builtin_prefetch(A + kt * 32 + 32, 0, 3);
    v16h a0 = load_a_rowmajor(A + kt * 32, D_);
    v16h a1 = load_a_rowmajor(A + (size_t)16 * D_ + kt * 32, D_);
    v16h bf[4];
#pragma unroll
    for (int ni = 0; ni < 4; ++ni)
      bf[ni] = load_b_nmajor(BT + (size_t)(ni * 16) * D_ + kt * 32, D_);
#pragma unroll
    for (int ni = 0; ni < 4; ++ni) {
      acc[0][ni] = wmma16(a0, bf[ni], acc[0][ni]);
      acc[1][ni] = wmma16(a1, bf[ni], acc[1][ni]);
    }
  }
  int lane = threadIdx.x & 31;
  int n = lane & 15, g = lane >> 4;
  float* out = outg + (size_t)b * W_ * W_;
#pragma unroll
  for (int mi = 0; mi < 2; ++mi)
#pragma unroll
    for (int ni = 0; ni < 4; ++ni)
#pragma unroll
      for (int r = 0; r < 8; ++r) {
        int row = wm * 32 + mi * 16 + g * 8 + r;
        int col = un * 64 + ni * 16 + n;
        float s = acc[mi][ni][r];
        if (MODE == 0) {
          out[(size_t)row * W_ + col] = s;
        } else {
          float sig = 1.0f / (1.0f + expf(-5.0f * s)) + 1e-5f;
          float sigma = expf(sig * 1.0986122886681098f) - 1.0f;  // 3^sig - 1
          float dist = fabsf((float)(row - col));
          float pr = (0.3989422804014327f / sigma) * expf(-dist * dist / (2.0f * sigma * sigma));
          out[(size_t)row * W_ + col] = pr;
        }
      }
}

// ---------------------------------------------------------------------------
// Wh = A[B*W x K] @ WT[N x K]^T + bias  -> f16 out; 32x64 blocked
// grid: ((B*W)/32, N/64)  block: 32
// ---------------------------------------------------------------------------
__global__ void k_whgemm(const half_t* __restrict__ A, const half_t* __restrict__ WT,
                         const float* __restrict__ bias, half_t* __restrict__ out,
                         int K, int N) {
  int mt = blockIdx.x, nt = blockIdx.y;
  const half_t* Ab = A + (size_t)(mt * 32) * K;
  const half_t* Bt = WT + (size_t)(nt * 64) * K;
  v8f acc[2][4] = {};
  for (int kt = 0; kt < K / 32; ++kt) {
    __builtin_prefetch(Ab + kt * 32 + 32, 0, 3);
    v16h a0 = load_a_rowmajor(Ab + kt * 32, K);
    v16h a1 = load_a_rowmajor(Ab + (size_t)16 * K + kt * 32, K);
    v16h bf[4];
#pragma unroll
    for (int ni = 0; ni < 4; ++ni)
      bf[ni] = load_b_nmajor(Bt + (size_t)(ni * 16) * K + kt * 32, K);
#pragma unroll
    for (int ni = 0; ni < 4; ++ni) {
      acc[0][ni] = wmma16(a0, bf[ni], acc[0][ni]);
      acc[1][ni] = wmma16(a1, bf[ni], acc[1][ni]);
    }
  }
  int lane = threadIdx.x & 31;
  int n = lane & 15, g = lane >> 4;
#pragma unroll
  for (int mi = 0; mi < 2; ++mi)
#pragma unroll
    for (int ni = 0; ni < 4; ++ni) {
      int col = nt * 64 + ni * 16 + n;
      float bo = bias[col];
#pragma unroll
      for (int r = 0; r < 8; ++r)
        out[(size_t)(mt * 32 + mi * 16 + g * 8 + r) * N + col] = (half_t)(acc[mi][ni][r] + bo);
    }
}

// e1 = Wh@ai + bai, e2 = Wh@aj + baj  (per-row dot products)
__global__ void k_evec(const half_t* __restrict__ Wh,
                       const float* __restrict__ ai, const float* __restrict__ bai,
                       const float* __restrict__ aj, const float* __restrict__ baj,
                       float* __restrict__ e1, float* __restrict__ e2, int K) {
  int row = blockIdx.x * blockDim.x + threadIdx.x;
  if (row >= B_ * W_) return;
  const half_t* r = Wh + (size_t)row * K;
  float s1 = 0.0f, s2 = 0.0f;
  for (int k = 0; k < K; ++k) {
    float v = (float)r[k];
    s1 += v * ai[k];
    s2 += v * aj[k];
  }
  e1[row] = s1 + bai[0];
  e2[row] = s2 + baj[0];
}

// att[row, :] = softmax_u( leakyrelu(e1[row] + e2[b,u]) )  -> f16
// (adj = scaled softmax scores, strictly > 0, so the mask is a no-op)
__global__ void k_att(const float* __restrict__ e1, const float* __restrict__ e2,
                      half_t* __restrict__ att) {
  int row = blockIdx.x;
  int b = row / W_;
  int t = threadIdx.x;
  float a1 = e1[row];
  const float* e2b = e2 + (size_t)b * W_;
  __shared__ float red[256];
  float x0 = a1 + e2b[t];
  float x1 = a1 + e2b[t + 256];
  x0 = x0 > 0.0f ? x0 : ALPHA_ * x0;
  x1 = x1 > 0.0f ? x1 : ALPHA_ * x1;
  red[t] = fmaxf(x0, x1);
  __syncthreads();
  for (int s = 128; s > 0; s >>= 1) { if (t < s) red[t] = fmaxf(red[t], red[t + s]); __syncthreads(); }
  float mx = red[0];
  __syncthreads();
  float ex0 = expf(x0 - mx), ex1 = expf(x1 - mx);
  red[t] = ex0 + ex1;
  __syncthreads();
  for (int s = 128; s > 0; s >>= 1) { if (t < s) red[t] += red[t + s]; __syncthreads(); }
  float inv = 1.0f / red[0];
  half_t* ar = att + (size_t)row * W_;
  ar[t]       = (half_t)(ex0 * inv);
  ar[t + 256] = (half_t)(ex1 * inv);
}

// ---------------------------------------------------------------------------
// out = elu( att[B: W x W] @ Wh )  with WhT [B: N x W] (N-major, contiguous K)
// grid: (W/32, N/64, B)  block: 32.  F32OUT: write f32 (layer2) else f16.
// ---------------------------------------------------------------------------
template <bool F32OUT>
__global__ void k_gatout(const half_t* __restrict__ att, const half_t* __restrict__ WhT,
                         half_t* __restrict__ outh, float* __restrict__ outf, int N) {
  int wm = blockIdx.x, nt = blockIdx.y, b = blockIdx.z;
  const half_t* A  = att + (size_t)b * W_ * W_ + (size_t)(wm * 32) * W_;
  const half_t* BT = WhT + (size_t)b * N * W_ + (size_t)(nt * 64) * W_;
  v8f acc[2][4] = {};
  for (int kt = 0; kt < W_ / 32; ++kt) {
    __builtin_prefetch(A + kt * 32 + 32, 0, 3);
    v16h a0 = load_a_rowmajor(A + kt * 32, W_);
    v16h a1 = load_a_rowmajor(A + (size_t)16 * W_ + kt * 32, W_);
    v16h bf[4];
#pragma unroll
    for (int ni = 0; ni < 4; ++ni)
      bf[ni] = load_b_nmajor(BT + (size_t)(ni * 16) * W_ + kt * 32, W_);
#pragma unroll
    for (int ni = 0; ni < 4; ++ni) {
      acc[0][ni] = wmma16(a0, bf[ni], acc[0][ni]);
      acc[1][ni] = wmma16(a1, bf[ni], acc[1][ni]);
    }
  }
  int lane = threadIdx.x & 31;
  int n = lane & 15, g = lane >> 4;
#pragma unroll
  for (int mi = 0; mi < 2; ++mi)
#pragma unroll
    for (int ni = 0; ni < 4; ++ni)
#pragma unroll
      for (int r = 0; r < 8; ++r) {
        float v = acc[mi][ni][r];
        v = v > 0.0f ? v : (expf(v) - 1.0f);  // elu
        size_t idx = ((size_t)b * W_ + wm * 32 + mi * 16 + g * 8 + r) * N + nt * 64 + ni * 16 + n;
        if (F32OUT) outf[idx] = v; else outh[idx] = (half_t)v;
      }
}

// in-place row softmax * 1/sqrt(D) on score region (rows of length W)
__global__ void k_scoresm(float* __restrict__ score) {
  int row = blockIdx.x;
  int t = threadIdx.x;
  float* r = score + (size_t)row * W_;
  __shared__ float red[256];
  float x0 = r[t], x1 = r[t + 256];
  red[t] = fmaxf(x0, x1);
  __syncthreads();
  for (int s = 128; s > 0; s >>= 1) { if (t < s) red[t] = fmaxf(red[t], red[t + s]); __syncthreads(); }
  float mx = red[0];
  __syncthreads();
  float ex0 = expf(x0 - mx), ex1 = expf(x1 - mx);
  red[t] = ex0 + ex1;
  __syncthreads();
  for (int s = 128; s > 0; s >>= 1) { if (t < s) red[t] += red[t + s]; __syncthreads(); }
  const float scale = 0.04419417382415922f;  // 1/sqrt(512)
  float inv = scale / red[0];
  r[t] = ex0 * inv;
  r[t + 256] = ex1 * inv;
}

// in-place row log_softmax on out region (rows of length D)
__global__ void k_logsm(float* __restrict__ out) {
  int row = blockIdx.x;
  int t = threadIdx.x;
  float* r = out + (size_t)row * D_;
  __shared__ float red[256];
  float x0 = r[t], x1 = r[t + 256];
  red[t] = fmaxf(x0, x1);
  __syncthreads();
  for (int s = 128; s > 0; s >>= 1) { if (t < s) red[t] = fmaxf(red[t], red[t + s]); __syncthreads(); }
  float mx = red[0];
  __syncthreads();
  red[t] = expf(x0 - mx) + expf(x1 - mx);
  __syncthreads();
  for (int s = 128; s > 0; s >>= 1) { if (t < s) red[t] += red[t + s]; __syncthreads(); }
  float ls = logf(red[0]);
  r[t]       = x0 - mx - ls;
  r[t + 256] = x1 - mx - ls;
}

// ---------------------------------------------------------------------------
extern "C" void kernel_launch(void* const* d_in, const int* in_sizes, int n_in,
                              void* d_out, int out_size, void* d_ws, size_t ws_size,
                              hipStream_t stream) {
  (void)in_sizes; (void)n_in; (void)out_size; (void)ws_size;
  const float* x    = (const float*)d_in[0];
  const float* wq   = (const float*)d_in[1];
  const float* bq   = (const float*)d_in[2];
  const float* wk   = (const float*)d_in[3];
  const float* bk   = (const float*)d_in[4];
  const float* wv   = (const float*)d_in[5];
  const float* bv   = (const float*)d_in[6];
  const float* wp   = (const float*)d_in[7];
  const float* bp   = (const float*)d_in[8];
  const float* W1   = (const float*)d_in[9];
  const float* b1   = (const float*)d_in[10];
  const float* ai1  = (const float*)d_in[11];
  const float* bai1 = (const float*)d_in[12];
  const float* aj1  = (const float*)d_in[13];
  const float* baj1 = (const float*)d_in[14];
  const float* W2   = (const float*)d_in[15];
  const float* b2   = (const float*)d_in[16];
  const float* ai2  = (const float*)d_in[17];
  const float* bai2 = (const float*)d_in[18];
  const float* aj2  = (const float*)d_in[19];
  const float* baj2 = (const float*)d_in[20];

  float* out   = (float*)d_out;                    // [B,W,D]
  float* score = out + (size_t)B_ * W_ * D_;       // [B,W,W]
  float* prior = score + (size_t)B_ * W_ * W_;     // [B,W,W]

  char* wsp = (char*)d_ws;
  auto alloc = [&](size_t bytes) -> void* {
    void* p = (void*)wsp;
    wsp += (bytes + 255) & ~(size_t)255;
    return p;
  };
  half_t* xh   = (half_t*)alloc((size_t)B_ * WP_ * D_ * 2);
  half_t* wt   = (half_t*)alloc((size_t)4 * 3 * D_ * D_ * 2);
  half_t* qh   = (half_t*)alloc((size_t)B_ * W_ * D_ * 2);
  half_t* kh   = (half_t*)alloc((size_t)B_ * W_ * D_ * 2);
  half_t* vh   = (half_t*)alloc((size_t)B_ * W_ * D_ * 2);
  half_t* ph   = (half_t*)alloc((size_t)B_ * W_ * D_ * 2);
  half_t* w1h  = (half_t*)alloc((size_t)H_ * D_ * 2);
  half_t* w2h  = (half_t*)alloc((size_t)D_ * H_ * 2);
  half_t* wh1  = (half_t*)alloc((size_t)B_ * W_ * H_ * 2);
  half_t* wh2  = (half_t*)alloc((size_t)B_ * W_ * D_ * 2);
  half_t* whT1 = (half_t*)alloc((size_t)B_ * W_ * H_ * 2);
  half_t* whT2 = (half_t*)alloc((size_t)B_ * W_ * D_ * 2);
  half_t* h1   = (half_t*)alloc((size_t)B_ * W_ * H_ * 2);
  half_t* atth = (half_t*)alloc((size_t)B_ * W_ * W_ * 2);
  float*  e1   = (float*)alloc((size_t)B_ * W_ * 4);
  float*  e2   = (float*)alloc((size_t)B_ * W_ * 4);

  dim3 blk32(32);

  // precision conversion / layout
  k_cvt_xpad<<<(B_ * WP_ * D_ + 255) / 256, 256, 0, stream>>>(x, xh);
  k_cvt_convw<<<(4 * 3 * D_ * D_ + 255) / 256, 256, 0, stream>>>(wq, wk, wv, wp, wt);
  k_cvt<<<(H_ * D_ + 255) / 256, 256, 0, stream>>>(W1, w1h, H_ * D_);
  k_cvt<<<(D_ * H_ + 255) / 256, 256, 0, stream>>>(W2, w2h, D_ * H_);

  // conv1d q,k,v,p (WMMA, 32x64 blocks, branch-free inner loop)
  k_conv<<<dim3(W_ / 32, D_ / 64, B_ * 4), blk32, 0, stream>>>(
      xh, wt, bq, bk, bv, bp, qh, kh, vh, ph);

  // score = softmax(q k^T) * scale
  k_qkgemm<0><<<dim3(W_ / 32, W_ / 64, B_), blk32, 0, stream>>>(qh, kh, score);
  k_scoresm<<<B_ * W_, 256, 0, stream>>>(score);

  // prior (WMMA + fused sigma/gaussian chain)
  k_qkgemm<1><<<dim3(W_ / 32, W_ / 64, B_), blk32, 0, stream>>>(ph, kh, prior);

  // GAT layer 1
  k_whgemm<<<dim3(B_ * W_ / 32, H_ / 64), blk32, 0, stream>>>(vh, w1h, b1, wh1, D_, H_);
  k_transpose<<<(B_ * W_ * H_ + 255) / 256, 256, 0, stream>>>(wh1, whT1, W_, H_);
  k_evec<<<(B_ * W_ + 255) / 256, 256, 0, stream>>>(wh1, ai1, bai1, aj1, baj1, e1, e2, H_);
  k_att<<<B_ * W_, 256, 0, stream>>>(e1, e2, atth);
  k_gatout<false><<<dim3(W_ / 32, H_ / 64, B_), blk32, 0, stream>>>(atth, whT1, h1, nullptr, H_);

  // GAT layer 2
  k_whgemm<<<dim3(B_ * W_ / 32, D_ / 64), blk32, 0, stream>>>(h1, w2h, b2, wh2, H_, D_);
  k_transpose<<<(B_ * W_ * D_ + 255) / 256, 256, 0, stream>>>(wh2, whT2, W_, D_);
  k_evec<<<(B_ * W_ + 255) / 256, 256, 0, stream>>>(wh2, ai2, bai2, aj2, baj2, e1, e2, D_);
  k_att<<<B_ * W_, 256, 0, stream>>>(e1, e2, atth);
  k_gatout<true><<<dim3(W_ / 32, D_ / 64, B_), blk32, 0, stream>>>(atth, whT2, nullptr, out, D_);

  // final log_softmax over D, in place
  k_logsm<<<B_ * W_, 256, 0, stream>>>(out);
}
